// TriangleAttentionEndingNode_23708219474365
// MI455X (gfx1250) — matrix operand order repaired and checked by reading, not compiled, not run on hardware
//
#include <hip/hip_runtime.h>
#include <stdint.h>

typedef __attribute__((ext_vector_type(16))) _Float16 v16h;
typedef __attribute__((ext_vector_type(8)))  float    v8f;

#define NRES   256
#define DMODEL 128
#define NHEAD  4
#define CHEAD  32
#define NPAIR  (NRES * NRES)              // 65536
#define QK_SCALE 0.17677669529663689f     // 1/sqrt(32)

// workspace byte offsets
#define WS_Q     (0u)
#define WS_K     (16u << 20)
#define WS_V     (32u << 20)
#define WS_G     (48u << 20)
#define WS_O     (64u << 20)
#define WS_BIAS  (80u << 20)              // 1 MB f32, transposed: [h][k*256+q]
#define WS_FQKV  (81u << 20)              // 96 KB  f16 swizzled qkv_w frags
#define WS_FGATE ((81u << 20) + 0x18000u) // 32 KB  f16 swizzled gate_w frags
#define WS_FO    ((81u << 20) + 0x20000u) // 32 KB  f16 swizzled o_w frags

// ---------------- WMMA fragment helpers (contiguous k = 16*(lane/16) + s, wave32) ---------

// A-matrix 16x32 f16 from LDS, row-major [16][stride_h] (stride even).
__device__ __forceinline__ v16h load_a_frag_lds(const _Float16* tile, int stride_h, int lane) {
  const int m  = lane & 15;
  const int kb = (lane < 16) ? 0 : 16;
  union { v16h v; uint32_t u[8]; } r;
  const uint32_t* row = (const uint32_t*)(tile + m * stride_h);
  #pragma unroll
  for (int t = 0; t < 4; ++t) r.u[t]     = row[(kb >> 1) + t];        // K = kb+2t, +1
  #pragma unroll
  for (int t = 0; t < 4; ++t) r.u[4 + t] = row[(kb >> 1) + 4 + t];    // K = kb+8+2t, +1
  return r.v;
}

// B-fragment from pre-swizzled weight buffer: one contiguous 32B load per lane.
__device__ __forceinline__ v16h load_b_frag_sw(const _Float16* F, int ct, int ks, int lane) {
  return *(const v16h*)(F + ((size_t)(((ct << 2) + ks) * 32 + lane) << 4));
}

// B = K^T tile (contraction over c=0..31) from LDS Kl[key][c] stride 32.
__device__ __forceinline__ v16h load_b_frag_kT(const _Float16* Kl, int key0, int lane) {
  const int n  = lane & 15;
  const int kb = (lane < 16) ? 0 : 16;
  union { v16h v; uint32_t u[8]; } r;
  const uint32_t* row = (const uint32_t*)(Kl + (key0 + n) * CHEAD);
  #pragma unroll
  for (int t = 0; t < 8; ++t) r.u[t] = row[(kb >> 1) + t];            // c = kb+2t, +1
  return r.v;
}

// B = V tile (contraction over 32 keys at key0) from transposed LDS Vt[c][key].
__device__ __forceinline__ v16h load_b_frag_v(const _Float16* Vt, int key0, int cc, int lane) {
  const int n  = lane & 15;
  const int kb = (lane < 16) ? 0 : 16;
  union { v16h v; uint32_t u[8]; } r;
  const uint32_t* row = (const uint32_t*)(Vt + (cc + n) * NRES + key0);
  #pragma unroll
  for (int t = 0; t < 8; ++t) r.u[t] = row[(kb >> 1) + t];            // key = key0+kb+2t, +1
  return r.v;
}

// ---------------- Kernel 0: one-time weight swizzle to B-frag layout ---------
__global__ __launch_bounds__(256) void k_swz_w(
    const float* __restrict__ qkv_w, const float* __restrict__ gate_w,
    const float* __restrict__ o_w,
    _Float16* __restrict__ fq, _Float16* __restrict__ fg, _Float16* __restrict__ fo)
{
  int id = blockIdx.x * 256 + threadIdx.x;          // 0..5119
  const float* W; _Float16* F; int ncols;
  if (id < 3072)      { W = qkv_w;  F = fq; ncols = 384; }
  else if (id < 4096) { W = gate_w; F = fg; ncols = 128; id -= 3072; }
  else                { W = o_w;    F = fo; ncols = 128; id -= 4096; }
  const int lane = id & 31;
  const int ks   = (id >> 5) & 3;
  const int ct   = id >> 7;
  const int n    = lane & 15;
  const int kb   = (lane < 16) ? 0 : 16;
  _Float16* dst  = F + ((size_t)id << 4);
  #pragma unroll
  for (int t = 0; t < 8; ++t) {
    dst[2 * t]     = (_Float16)W[(size_t)(ks * 32 + kb + 2 * t)     * ncols + ct * 16 + n];
    dst[2 * t + 1] = (_Float16)W[(size_t)(ks * 32 + kb + 2 * t + 1) * ncols + ct * 16 + n];
  }
}

// ---------------- Kernel A: LayerNorm + qkv/gate/bias projections ------------
__global__ __launch_bounds__(256) void k_ln_proj(
    const float* __restrict__ Zraw, const float* __restrict__ ln_g,
    const float* __restrict__ ln_b, const float* __restrict__ b_w,
    const _Float16* __restrict__ fq, const _Float16* __restrict__ fg,
    const float* __restrict__ gate_b,
    _Float16* __restrict__ qw, _Float16* __restrict__ kw,
    _Float16* __restrict__ vw, _Float16* __restrict__ gw,
    float* __restrict__ biasT)
{
  __shared__ _Float16 sA[16][DMODEL];
  const int t    = threadIdx.x;
  const int lane = t & 31;
  const int wid  = t >> 5;
  const int r    = t >> 4;
  const int cs   = t & 15;
  const int p0   = blockIdx.x * 16;
  const int p    = p0 + r;
  const int j    = p >> 8, i = p & 255;
  const float* zrow = Zraw + ((size_t)i * NRES + j) * DMODEL;   // transposed read
  const int c0 = cs * 8;

  float x[8];
  { const float4 a = *(const float4*)(zrow + c0);
    const float4 b = *(const float4*)(zrow + c0 + 4);
    x[0]=a.x; x[1]=a.y; x[2]=a.z; x[3]=a.w; x[4]=b.x; x[5]=b.y; x[6]=b.z; x[7]=b.w; }
  float s1 = 0.f, s2 = 0.f;
  #pragma unroll
  for (int q = 0; q < 8; ++q) { s1 += x[q]; s2 += x[q] * x[q]; }
  #pragma unroll
  for (int msk = 1; msk < 16; msk <<= 1) {
    s1 += __shfl_xor(s1, msk, 32); s2 += __shfl_xor(s2, msk, 32);
  }
  const float mu   = s1 * (1.f / 128.f);
  const float var  = s2 * (1.f / 128.f) - mu * mu;
  const float rstd = rsqrtf(var + 1e-5f);

  float g8[8], b8[8];
  { const float4 a = *(const float4*)(ln_g + c0); const float4 b = *(const float4*)(ln_g + c0 + 4);
    g8[0]=a.x; g8[1]=a.y; g8[2]=a.z; g8[3]=a.w; g8[4]=b.x; g8[5]=b.y; g8[6]=b.z; g8[7]=b.w; }
  { const float4 a = *(const float4*)(ln_b + c0); const float4 b = *(const float4*)(ln_b + c0 + 4);
    b8[0]=a.x; b8[1]=a.y; b8[2]=a.z; b8[3]=a.w; b8[4]=b.x; b8[5]=b.y; b8[6]=b.z; b8[7]=b.w; }

  float bh[4] = {0.f, 0.f, 0.f, 0.f};
  #pragma unroll
  for (int q = 0; q < 8; ++q) {
    const int c = c0 + q;
    const float zn = (x[q] - mu) * rstd * g8[q] + b8[q];
    sA[r][c] = (_Float16)zn;
    const float4 bw = *(const float4*)(b_w + c * 4);              // all 4 heads
    bh[0] += zn * bw.x; bh[1] += zn * bw.y; bh[2] += zn * bw.z; bh[3] += zn * bw.w;
  }
  #pragma unroll
  for (int h = 0; h < 4; ++h)
    #pragma unroll
    for (int msk = 1; msk < 16; msk <<= 1) bh[h] += __shfl_xor(bh[h], msk, 32);
  // transposed store: biasT[h][k*256 + q] with (q,k) = (j,i) for this row
  if (cs < 4) biasT[(size_t)cs * NPAIR + (size_t)i * NRES + j] = bh[cs];

  __syncthreads();

  #pragma unroll
  for (int tt = 0; tt < 4; ++tt) {
    const int ct = wid * 4 + tt;                 // wave-uniform
    const _Float16* F; int ctl, col0;
    if (ct < 24) { F = fq; ctl = ct;      col0 = ct * 16; }
    else         { F = fg; ctl = ct - 24; col0 = (ct - 24) * 16; }
    v8f acc = {};
    #pragma unroll
    for (int ks = 0; ks < 4; ++ks) {
      v16h a = load_a_frag_lds(&sA[0][ks * 32], DMODEL, lane);
      v16h b = load_b_frag_sw(F, ctl, ks, lane);
      acc = __builtin_amdgcn_wmma_f32_16x16x32_f16(false, a, false, b, (short)0, acc, false, false);
    }
    const int n  = lane & 15;
    const int mb = (lane < 16) ? 0 : 8;
    const int col = col0 + n;
    const size_t base = (size_t)(p0 + mb) * DMODEL;
    // hoisted wave-uniform branch -> 4 straight-line store loops
    if (ct < 8) {
      #pragma unroll
      for (int v = 0; v < 8; ++v)
        qw[base + (size_t)v * DMODEL + col] = (_Float16)(acc[v] * QK_SCALE);
    } else if (ct < 16) {
      #pragma unroll
      for (int v = 0; v < 8; ++v)
        kw[base + (size_t)v * DMODEL + col - 128] = (_Float16)acc[v];
    } else if (ct < 24) {
      #pragma unroll
      for (int v = 0; v < 8; ++v)
        vw[base + (size_t)v * DMODEL + col - 256] = (_Float16)acc[v];
    } else {
      const float gb = gate_b[col];
      #pragma unroll
      for (int v = 0; v < 8; ++v) {
        const float g = __builtin_amdgcn_rcpf(1.f + __expf(-(acc[v] + gb)));
        gw[base + (size_t)v * DMODEL + col] = (_Float16)g;
      }
    }
  }
}

// ---------------- Kernel B: attention per (j, h), flash-style softmax --------
__global__ __launch_bounds__(256) void k_attn(
    const _Float16* __restrict__ qw, const _Float16* __restrict__ kw,
    const _Float16* __restrict__ vw, const float* __restrict__ biasT,
    _Float16* __restrict__ ow)
{
  __shared__ _Float16 Kl[NRES * CHEAD];   // [key][c]
  __shared__ _Float16 Vt[CHEAD * NRES];   // [c][key] transposed
  __shared__ _Float16 Pl[8][16 * 32];     // per-wave P tile

  const int t = threadIdx.x, lane = t & 31, wid = t >> 5;
  const int j = blockIdx.z, h = blockIdx.y;
  const int i0 = blockIdx.x * 128 + wid * 16;
  const size_t hoff = (size_t)h * CHEAD;

  { // stage K (row-major) and V (transposed)
    const int rr = t;
    const uint4* k4 = (const uint4*)(kw + ((size_t)(j * NRES + rr)) * DMODEL + hoff);
    uint4* kd = (uint4*)(Kl + rr * CHEAD);
    #pragma unroll
    for (int q = 0; q < 4; ++q) kd[q] = k4[q];
    const uint4* v4 = (const uint4*)(vw + ((size_t)(j * NRES + rr)) * DMODEL + hoff);
    #pragma unroll
    for (int q = 0; q < 4; ++q) {
      union { uint4 u; _Float16 hv[8]; } vv; vv.u = v4[q];
      #pragma unroll
      for (int e = 0; e < 8; ++e) Vt[(q * 8 + e) * NRES + rr] = vv.hv[e];
    }
  }
  __syncthreads();

  const int m  = lane & 15;
  const int kb = (lane < 16) ? 0 : 16;
  union { v16h v; uint32_t u[8]; } qf;
  {
    const uint32_t* qrow = (const uint32_t*)(qw + ((size_t)(j * NRES + i0 + m)) * DMODEL + hoff);
    #pragma unroll
    for (int q = 0; q < 4; ++q) qf.u[q]     = qrow[(kb >> 1) + q];
    #pragma unroll
    for (int q = 0; q < 4; ++q) qf.u[4 + q] = qrow[(kb >> 1) + 4 + q];
  }

  const int n  = lane & 15;
  const int mb = (lane < 16) ? 0 : 8;
  float mrun[8], lrun[8];
  #pragma unroll
  for (int v = 0; v < 8; ++v) { mrun[v] = -1e30f; lrun[v] = 0.f; }
  v8f acc0 = {}, acc1 = {};
  const float* bh = biasT + (size_t)h * NPAIR + i0 + mb;   // + key*256 below

  for (int s = 0; s < 8; ++s) {
    const int key0 = s * 32;
    v8f z0 = {}, z1 = {};
    v16h bk0 = load_b_frag_kT(Kl, key0, lane);
    v8f S0 = __builtin_amdgcn_wmma_f32_16x16x32_f16(false, qf.v, false, bk0, (short)0, z0, false, false);
    v16h bk1 = load_b_frag_kT(Kl, key0 + 16, lane);
    v8f S1 = __builtin_amdgcn_wmma_f32_16x16x32_f16(false, qf.v, false, bk1, (short)0, z1, false, false);

    // pair bias: transposed layout -> 8 contiguous floats per lane per tile
    float bb0[8], bb1[8];
    { const float* bp = bh + (size_t)(key0 + n) * NRES;
      const float4 a = *(const float4*)bp; const float4 b = *(const float4*)(bp + 4);
      bb0[0]=a.x; bb0[1]=a.y; bb0[2]=a.z; bb0[3]=a.w; bb0[4]=b.x; bb0[5]=b.y; bb0[6]=b.z; bb0[7]=b.w; }
    { const float* bp = bh + (size_t)(key0 + 16 + n) * NRES;
      const float4 a = *(const float4*)bp; const float4 b = *(const float4*)(bp + 4);
      bb1[0]=a.x; bb1[1]=a.y; bb1[2]=a.z; bb1[3]=a.w; bb1[4]=b.x; bb1[5]=b.y; bb1[6]=b.z; bb1[7]=b.w; }

    float e0[8], e1[8], mt[8];
    #pragma unroll
    for (int v = 0; v < 8; ++v) {
      e0[v] = S0[v] + bb0[v];
      e1[v] = S1[v] + bb1[v];
      mt[v] = fmaxf(e0[v], e1[v]);
    }
    #pragma unroll
    for (int v = 0; v < 8; ++v)
      #pragma unroll
      for (int msk = 1; msk < 16; msk <<= 1) mt[v] = fmaxf(mt[v], __shfl_xor(mt[v], msk, 32));

    float alpha[8], rs[8];
    #pragma unroll
    for (int v = 0; v < 8; ++v) {
      const float mn = fmaxf(mrun[v], mt[v]);
      alpha[v] = __expf(mrun[v] - mn);
      mrun[v] = mn;
      e0[v] = __expf(e0[v] - mn);
      e1[v] = __expf(e1[v] - mn);
      rs[v] = e0[v] + e1[v];
    }
    #pragma unroll
    for (int v = 0; v < 8; ++v)
      #pragma unroll
      for (int msk = 1; msk < 16; msk <<= 1) rs[v] += __shfl_xor(rs[v], msk, 32);
    #pragma unroll
    for (int v = 0; v < 8; ++v) {
      lrun[v] = lrun[v] * alpha[v] + rs[v];
      acc0[v] *= alpha[v];
      acc1[v] *= alpha[v];
    }

    _Float16* P = &Pl[wid][0];
    #pragma unroll
    for (int v = 0; v < 8; ++v) {
      P[(mb + v) * 32 + n]      = (_Float16)e0[v];
      P[(mb + v) * 32 + 16 + n] = (_Float16)e1[v];
    }
    v16h pa  = load_a_frag_lds(P, 32, lane);
    v16h bv0 = load_b_frag_v(Vt, key0, 0, lane);
    acc0 = __builtin_amdgcn_wmma_f32_16x16x32_f16(false, pa, false, bv0, (short)0, acc0, false, false);
    v16h bv1 = load_b_frag_v(Vt, key0, 16, lane);
    acc1 = __builtin_amdgcn_wmma_f32_16x16x32_f16(false, pa, false, bv1, (short)0, acc1, false, false);
  }

  #pragma unroll
  for (int v = 0; v < 8; ++v) {
    const float inv = __builtin_amdgcn_rcpf(lrun[v]);
    const size_t off = ((size_t)(j * NRES + i0 + mb + v)) * DMODEL + hoff;
    ow[off + n]      = (_Float16)(acc0[v] * inv);
    ow[off + 16 + n] = (_Float16)(acc1[v] * inv);
  }
}

// ---------------- Kernel C: gated output projection + residual ---------------
__global__ __launch_bounds__(256) void k_out(
    const float* __restrict__ Zraw, const _Float16* __restrict__ ow,
    const _Float16* __restrict__ gw, const _Float16* __restrict__ fo,
    const float* __restrict__ out_bias, float* __restrict__ out)
{
  __shared__ _Float16 sA[16][DMODEL];
  const int t = threadIdx.x, lane = t & 31, wid = t >> 5;
  const int r = t >> 4, cs = t & 15;
  const int p0 = blockIdx.x * 16;
  const int c0 = cs * 8;
  {
    const size_t off = (size_t)(p0 + r) * DMODEL + c0;
    union { uint4 u; _Float16 hv[8]; } o4, g4;
    o4.u = *(const uint4*)(ow + off);
    g4.u = *(const uint4*)(gw + off);
    #pragma unroll
    for (int e = 0; e < 8; ++e)
      sA[r][c0 + e] = (_Float16)((float)o4.hv[e] * (float)g4.hv[e]);
  }
  __syncthreads();

  const int col0 = wid * 16;
  v8f acc = {};
  #pragma unroll
  for (int ks = 0; ks < 4; ++ks) {
    v16h a = load_a_frag_lds(&sA[0][ks * 32], DMODEL, lane);
    v16h b = load_b_frag_sw(fo, wid, ks, lane);
    acc = __builtin_amdgcn_wmma_f32_16x16x32_f16(false, a, false, b, (short)0, acc, false, false);
  }
  const int n = lane & 15, mb = (lane < 16) ? 0 : 8;
  const int d = col0 + n;
  const float ob = out_bias[d];
  #pragma unroll
  for (int v = 0; v < 8; ++v) {
    const int pp = p0 + mb + v;
    const int jj = pp >> 8, ii = pp & 255;
    const size_t zoff = ((size_t)ii * NRES + jj) * DMODEL + d;   // transpose back
    out[zoff] = Zraw[zoff] + acc[v] + ob;
  }
}

// ---------------- launch -----------------------------------------------------
extern "C" void kernel_launch(void* const* d_in, const int* in_sizes, int n_in,
                              void* d_out, int out_size, void* d_ws, size_t ws_size,
                              hipStream_t stream) {
  const float* Zraw   = (const float*)d_in[0];
  const float* ln_g   = (const float*)d_in[1];
  const float* ln_b   = (const float*)d_in[2];
  const float* b_w    = (const float*)d_in[3];
  const float* qkv_w  = (const float*)d_in[4];
  const float* gate_w = (const float*)d_in[5];
  const float* gate_b = (const float*)d_in[6];
  const float* o_w    = (const float*)d_in[7];
  const float* obias  = (const float*)d_in[8];

  char* ws = (char*)d_ws;
  _Float16* qw    = (_Float16*)(ws + WS_Q);
  _Float16* kw    = (_Float16*)(ws + WS_K);
  _Float16* vw    = (_Float16*)(ws + WS_V);
  _Float16* gw    = (_Float16*)(ws + WS_G);
  _Float16* owk   = (_Float16*)(ws + WS_O);
  float*    biasT = (float*)(ws + WS_BIAS);
  _Float16* fq    = (_Float16*)(ws + WS_FQKV);
  _Float16* fg    = (_Float16*)(ws + WS_FGATE);
  _Float16* fo    = (_Float16*)(ws + WS_FO);

  k_swz_w<<<20, 256, 0, stream>>>(qkv_w, gate_w, o_w, fq, fg, fo);
  k_ln_proj<<<NPAIR / 16, 256, 0, stream>>>(Zraw, ln_g, ln_b, b_w, fq, fg,
                                            gate_b, qw, kw, vw, gw, biasT);
  dim3 gB(2, NHEAD, NRES);
  k_attn<<<gB, 256, 0, stream>>>(qw, kw, vw, biasT, owk);
  k_out<<<NPAIR / 16, 256, 0, stream>>>(Zraw, owk, gw, fo, obias, (float*)d_out);
}
